// NeuralODE_51196010169010
// MI455X (gfx1250) — matrix operand order repaired and checked by reading, not compiled, not run on hardware
//
#include <hip/hip_runtime.h>

// ---------------------------------------------------------------------------
// Neural ODE (dopri5, fixed 512 steps) for a 3->256->256->3 tanh MLP field.
// MI455X / gfx1250: f16 WMMA (v_wmma_f32_16x16x32_f16) with fp32 accumulate,
// W2 resident in LDS as a transposed f16 panel, whole integration on-chip.
// ---------------------------------------------------------------------------

typedef __attribute__((ext_vector_type(16))) _Float16 v16h;
typedef __attribute__((ext_vector_type(8)))  _Float16 v8h;
typedef __attribute__((ext_vector_type(4)))  _Float16 v4h;
typedef __attribute__((ext_vector_type(2)))  _Float16 v2h;
typedef __attribute__((ext_vector_type(8)))  float    v8f;

#define HID      256
#define NSTEPS   512
#define THREADS  256         // 8 wave32
#define SPW      32          // samples per wave (2 row-blocks of 16)
#define SPB      256         // samples per block
#define NBLK     128         // 32768 / 256

// f16 staging / weight panel stride: 272 halves = 544 B (=> 8-dword bank skew,
// conflict-free 16-lane b128 fragment loads)
#define LSTR     272

// LDS layout (bytes)
#define W2T_OFF  0                         // [256][272] f16  = 139264
#define W3T_OFF  139264                    // [16][272]  f16  = 8704
#define B2_OFF   147968                    // [256] f32       = 1024
#define HS_OFF   148992                    // per wave: [32][272] f16 = 17408
#define HS_BYTES 17408
#define FST_OFF  (148992 + 8 * 17408)      // per wave: [32][4] f32 = 512
#define SMEM_BYTES (FST_OFF + 8 * 512)     // = 292352 (< 320KB WGP LDS)

// Dormand-Prince 5(4) tableau
#define C_A21  (1.0f/5.0f)
#define C_A31  (3.0f/40.0f)
#define C_A32  (9.0f/40.0f)
#define C_A41  (44.0f/45.0f)
#define C_A42  (-56.0f/15.0f)
#define C_A43  (32.0f/9.0f)
#define C_A51  (19372.0f/6561.0f)
#define C_A52  (-25360.0f/2187.0f)
#define C_A53  (64448.0f/6561.0f)
#define C_A54  (-212.0f/729.0f)
#define C_A61  (9017.0f/3168.0f)
#define C_A62  (-355.0f/33.0f)
#define C_A63  (46732.0f/5247.0f)
#define C_A64  (49.0f/176.0f)
#define C_A65  (-5103.0f/18656.0f)
#define C_B1   (35.0f/384.0f)
#define C_B3   (500.0f/1113.0f)
#define C_B4   (125.0f/192.0f)
#define C_B5   (-2187.0f/6784.0f)
#define C_B6   (11.0f/84.0f)

__device__ __forceinline__ float fast_tanh(float x) {
#if __has_builtin(__builtin_amdgcn_tanhf)
  return __builtin_amdgcn_tanhf(x);            // v_tanh_f32 (gfx1250 TRANS op)
#elif __has_builtin(__builtin_amdgcn_tanh_f32)
  return __builtin_amdgcn_tanh_f32(x);
#else
  float e = __expf(2.0f * x);
  return 1.0f - 2.0f / (e + 1.0f);
#endif
}

__device__ __forceinline__ float bcast_lane(float v, int src) {
  return __int_as_float(__builtin_amdgcn_readlane(__float_as_int(v), src));
}

// v_cvt_pk_f16_f32-style pack; builtin returns __fp16x2 -> bit-cast to v2h.
__device__ __forceinline__ v2h pack2(float a, float b) {
  return __builtin_bit_cast(v2h, __builtin_amdgcn_cvt_pkrtz(a, b));
}

__device__ __forceinline__ v8f wmma16(v16h a, v16h b, v8f c) {
  return __builtin_amdgcn_wmma_f32_16x16x32_f16(
      /*neg_a=*/false, a, /*neg_b=*/false, b,
      /*c_mod=*/(short)0, c, /*reuse_a=*/false, /*reuse_b=*/false);
}

// Build a 16-half fragment from two contiguous 16-byte LDS chunks.
__device__ __forceinline__ v16h frag16(const _Float16* p, int hi_off) {
  v8h lo = *(const v8h*)p;
  v8h hi = *(const v8h*)(p + hi_off);
  return __builtin_shufflevector(lo, hi, 0,1,2,3,4,5,6,7,8,9,10,11,12,13,14,15);
}

extern "C" __global__ void __launch_bounds__(THREADS)
node_dopri5_wmma(const float* __restrict__ y0g,
                 const float* __restrict__ W1g,
                 const float* __restrict__ b1g,
                 const float* __restrict__ W2g,
                 const float* __restrict__ b2g,
                 const float* __restrict__ W3g,
                 const float* __restrict__ b3g,
                 float* __restrict__ outg) {
  extern __shared__ char smem[];
  _Float16* W2T = (_Float16*)(smem + W2T_OFF);   // [n][k] = W2[k][n], f16
  _Float16* W3T = (_Float16*)(smem + W3T_OFF);   // [j pad 16][k], f16
  float*    b2s = (float*)(smem + B2_OFF);

  const int tid  = threadIdx.x;
  const int lane = tid & 31;
  const int wave = tid >> 5;
  const int mr   = lane & 15;        // fragment row / column-in-tile
  const int kh   = lane >> 4;        // K half selector (WMMA lane layout)

  _Float16* hs  = (_Float16*)(smem + HS_OFF + wave * HS_BYTES); // [32][272] f16
  float*    fst = (float*)(smem + FST_OFF + wave * 512);        // [32][4] f32

  // ---- one-time weight prep (cooperative) --------------------------------
  for (int i = tid; i < HID * HID; i += THREADS) {
    int n = i & 255, k = i >> 8;
    W2T[n * LSTR + k] = (_Float16)W2g[k * HID + n];   // transpose + f16
  }
  for (int i = tid; i < 16 * HID; i += THREADS) {
    int n = i & 15, k = i >> 4;
    float v = (n < 3) ? W3g[k * 3 + n] : 0.0f;        // pad N: 3 -> 16
    W3T[n * LSTR + k] = (_Float16)v;
  }
  if (tid < HID) b2s[tid] = b2g[tid];
  __syncthreads();

  // ---- per-lane constants -------------------------------------------------
  float w1r[3][8], b1r[8];
#pragma unroll
  for (int d = 0; d < 3; ++d)
#pragma unroll
    for (int t = 0; t < 8; ++t) w1r[d][t] = W1g[d * HID + lane * 8 + t];
#pragma unroll
  for (int t = 0; t < 8; ++t) b1r[t] = b1g[lane * 8 + t];
  const float b3r0 = b3g[0], b3r1 = b3g[1], b3r2 = b3g[2];

  const int sample = blockIdx.x * SPB + wave * SPW + lane; // lane owns 1 sample
  float y[3];
#pragma unroll
  for (int j = 0; j < 3; ++j) y[j] = y0g[sample * 3 + j];

  // ---- vector field: fo = W3^T tanh(W2^T tanh(W1^T yt + b1) + b2) + b3 ----
  auto feval = [&](float yt0, float yt1, float yt2, float* fo) {
    // layer 1 (K=3, VALU) -> h1 staged as f16 [32][LSTR]
#pragma unroll 1
    for (int m = 0; m < SPW; ++m) {
      float a = bcast_lane(yt0, m);
      float b = bcast_lane(yt1, m);
      float c = bcast_lane(yt2, m);
      v2h pk[4];
#pragma unroll
      for (int t = 0; t < 8; t += 2) {
        float z0 = fmaf(a, w1r[0][t],
                   fmaf(b, w1r[1][t], fmaf(c, w1r[2][t], b1r[t])));
        float z1 = fmaf(a, w1r[0][t + 1],
                   fmaf(b, w1r[1][t + 1], fmaf(c, w1r[2][t + 1], b1r[t + 1])));
        pk[t >> 1] = pack2(fast_tanh(z0), fast_tanh(z1));
      }
      v4h q0 = __builtin_shufflevector(pk[0], pk[1], 0, 1, 2, 3);
      v4h q1 = __builtin_shufflevector(pk[2], pk[3], 0, 1, 2, 3);
      v8h hv = __builtin_shufflevector(q0, q1, 0, 1, 2, 3, 4, 5, 6, 7);
      *(v8h*)&hs[m * LSTR + lane * 8] = hv;   // 16B store, lane-contiguous k
    }
    __syncthreads();

    // A fragments for both 16-row blocks, all 8 K-blocks (register resident)
    v16h A[2][8];
#pragma unroll
    for (int rb = 0; rb < 2; ++rb)
#pragma unroll
      for (int kb = 0; kb < 8; ++kb)
        A[rb][kb] = frag16(hs + (rb * 16 + mr) * LSTR + kb * 32 + kh * 8, 16);

    // layer 2 (256x256 GEMM): 16 N-tiles x 8 K-steps x 2 row blocks of WMMA
#pragma unroll 1
    for (int tile = 0; tile < 16; ++tile) {
      v8f acc0 = {}; v8f acc1 = {};
      const _Float16* bp = W2T + (tile * 16 + mr) * LSTR + kh * 8;
#pragma unroll
      for (int kb = 0; kb < 8; ++kb) {
        v16h Bf = frag16(bp + kb * 32, 16);
        acc0 = wmma16(A[0][kb], Bf, acc0);
        acc1 = wmma16(A[1][kb], Bf, acc1);
      }
      float bn = b2s[tile * 16 + mr];
#pragma unroll
      for (int r = 0; r < 8; ++r) {
        float h0 = fast_tanh(acc0[r] + bn);
        float h1 = fast_tanh(acc1[r] + bn);
        hs[(kh * 8 + r) * LSTR + tile * 16 + mr]        = (_Float16)h0;  // rb0
        hs[(16 + kh * 8 + r) * LSTR + tile * 16 + mr]   = (_Float16)h1;  // rb1
      }
    }
    __syncthreads();

    // layer 3 (256 -> 3, N padded to 16): 16 WMMAs against W3 panel
    v8f o0 = {}; v8f o1 = {};
    const _Float16* wp = W3T + mr * LSTR + kh * 8;
#pragma unroll
    for (int kb = 0; kb < 8; ++kb) {
      v16h a0 = frag16(hs + (0  + mr) * LSTR + kb * 32 + kh * 8, 16);
      v16h a1 = frag16(hs + (16 + mr) * LSTR + kb * 32 + kh * 8, 16);
      v16h Bf = frag16(wp + kb * 32, 16);
      o0 = wmma16(a0, Bf, o0);
      o1 = wmma16(a1, Bf, o1);
    }
    // scatter D-tile (M=r+8*kh, N=mr) back to sample-owner lanes via LDS
    if (mr < 3) {
#pragma unroll
      for (int r = 0; r < 8; ++r) {
        fst[(kh * 8 + r) * 4 + mr]      = o0[r];
        fst[(16 + kh * 8 + r) * 4 + mr] = o1[r];
      }
    }
    __syncthreads();
    fo[0] = fst[lane * 4 + 0] + b3r0;
    fo[1] = fst[lane * 4 + 1] + b3r1;
    fo[2] = fst[lane * 4 + 2] + b3r2;
  };

  // ---- dopri5 fixed grid --------------------------------------------------
  const float hstep = 10.0f / (float)NSTEPS;
  float k1[3], k2[3], k3[3], k4[3], k5[3], k6[3], yt[3];

#pragma unroll 1
  for (int step = 0; step < NSTEPS; ++step) {
    feval(y[0], y[1], y[2], k1);

#pragma unroll
    for (int j = 0; j < 3; ++j) yt[j] = y[j] + hstep * (C_A21 * k1[j]);
    feval(yt[0], yt[1], yt[2], k2);

#pragma unroll
    for (int j = 0; j < 3; ++j)
      yt[j] = y[j] + hstep * (C_A31 * k1[j] + C_A32 * k2[j]);
    feval(yt[0], yt[1], yt[2], k3);

#pragma unroll
    for (int j = 0; j < 3; ++j)
      yt[j] = y[j] + hstep * (C_A41 * k1[j] + C_A42 * k2[j] + C_A43 * k3[j]);
    feval(yt[0], yt[1], yt[2], k4);

#pragma unroll
    for (int j = 0; j < 3; ++j)
      yt[j] = y[j] + hstep * (C_A51 * k1[j] + C_A52 * k2[j] +
                              C_A53 * k3[j] + C_A54 * k4[j]);
    feval(yt[0], yt[1], yt[2], k5);

#pragma unroll
    for (int j = 0; j < 3; ++j)
      yt[j] = y[j] + hstep * (C_A61 * k1[j] + C_A62 * k2[j] + C_A63 * k3[j] +
                              C_A64 * k4[j] + C_A65 * k5[j]);
    feval(yt[0], yt[1], yt[2], k6);

#pragma unroll
    for (int j = 0; j < 3; ++j)
      y[j] += hstep * (C_B1 * k1[j] + C_B3 * k3[j] + C_B4 * k4[j] +
                       C_B5 * k5[j] + C_B6 * k6[j]);
  }

#pragma unroll
  for (int j = 0; j < 3; ++j) outg[sample * 3 + j] = y[j];
}

extern "C" void kernel_launch(void* const* d_in, const int* in_sizes, int n_in,
                              void* d_out, int out_size, void* d_ws,
                              size_t ws_size, hipStream_t stream) {
  (void)in_sizes; (void)n_in; (void)out_size; (void)d_ws; (void)ws_size;
  // setup_inputs order: ts, y0, W1, b1, W2, b2, W3, b3 (all fp32)
  const float* y0 = (const float*)d_in[1];
  const float* W1 = (const float*)d_in[2];
  const float* b1 = (const float*)d_in[3];
  const float* W2 = (const float*)d_in[4];
  const float* b2 = (const float*)d_in[5];
  const float* W3 = (const float*)d_in[6];
  const float* b3 = (const float*)d_in[7];
  float* out = (float*)d_out;

  hipLaunchKernelGGL(node_dopri5_wmma, dim3(NBLK), dim3(THREADS),
                     SMEM_BYTES, stream, y0, W1, b1, W2, b2, W3, b3, out);
}
